// Latent_ODE_56856777065067
// MI455X (gfx1250) — matrix operand order repaired and checked by reading, not compiled, not run on hardware
//
#include <hip/hip_runtime.h>
#include <hip/hip_bf16.h>

typedef __attribute__((ext_vector_type(16))) _Float16 v16h;
typedef __attribute__((ext_vector_type(8)))  float    v8f;

#define T_IN   512
#define T_OUT  1024
#define NB     256
#define OBS    192
#define RHID   20
#define NHID   20
#define LATENT 4
#define GDIM   60   // 3*RHID

// ---------------------------------------------------------------------------
// Kernel 1: xW = flip_t(x).reshape(B,T,192) @ W_ih^T + b_ih   -> ws [B*T, 64]
// WMMA f16 GEMM, M=131072, K=192 (6 k-tiles), N=60 padded to 64 (4 n-tiles).
// W_ih staged in LDS padded to 64 rows so every B-fragment is one aligned
// 16-half contiguous load (ds_load_b128), no divergent guards.
// ---------------------------------------------------------------------------
__global__ __launch_bounds__(256) void k_xw_gemm(const float* __restrict__ x,
                                                 const float* __restrict__ W_ih,
                                                 const float* __restrict__ b_ih,
                                                 float* __restrict__ xW) {
    __shared__ _Float16 wlds[64 * OBS];            // padded rows 60..63 = 0
    __shared__ float    biasp[64];
    for (int i = threadIdx.x; i < 64 * OBS; i += 256) {
        const int n = i / OBS, k = i - n * OBS;
        wlds[i] = (n < GDIM) ? (_Float16)W_ih[n * OBS + k] : (_Float16)0.f;
    }
    if (threadIdx.x < 64)
        biasp[threadIdx.x] = (threadIdx.x < GDIM) ? b_ih[threadIdx.x] : 0.f;
    __syncthreads();

    const int wave  = threadIdx.x >> 5;
    const int lane  = threadIdx.x & 31;
    const int rowl  = lane & 15;
    const int hi    = lane >> 4;                   // lane group 0/1
    const int mtile = blockIdx.x * 8 + wave;
    const int r     = mtile * 16 + rowl;           // row = b*512 + t (flipped t)
    const int b     = r >> 9;
    const int t     = r & 511;
    const float* xrow = x + (size_t)b * 3 * (T_IN * 64) + (size_t)(511 - t) * 64;

    v8f acc[4];
    #pragma unroll
    for (int nt = 0; nt < 4; ++nt) { v8f z = {}; acc[nt] = z; }

    #pragma unroll
    for (int kt = 0; kt < 6; ++kt) {
        const int ks = kt * 32;
        // A fragment: lane group hi covers K = ks+hi*16+{0..15}
        v16h a;
        #pragma unroll
        for (int g = 0; g < 2; ++g) {
            const int kk0 = ks + hi * 16 + g * 8;
            const int c   = kk0 >> 6;              // channel (k = c*64 + f)
            const int f0  = kk0 & 63;              // 8-aligned: never crosses 64
            const float* p = xrow + (size_t)c * (T_IN * 64) + f0;
            #pragma unroll
            for (int u = 0; u < 8; ++u) a[g * 8 + u] = (_Float16)p[u];
        }
        #pragma unroll
        for (int nt = 0; nt < 4; ++nt) {
            const int n = nt * 16 + rowl;          // padded: always valid
            const v16h bf = *(const v16h*)&wlds[n * OBS + ks + hi * 16];
            acc[nt] = __builtin_amdgcn_wmma_f32_16x16x32_f16(
                false, a, false, bf, (short)0, acc[nt], false, false);
        }
    }

    #pragma unroll
    for (int nt = 0; nt < 4; ++nt) {
        const int   n    = nt * 16 + rowl;
        const float bias = biasp[n];
        #pragma unroll
        for (int v = 0; v < 8; ++v) {
            const int rr = mtile * 16 + v + hi * 8;
            xW[(size_t)rr * 64 + n] = acc[nt][v] + bias;
        }
    }
}

// ---------------------------------------------------------------------------
// Kernel 2: GRU scan over 512 steps. 8 batch elements per 256-thread block,
// one wave each. W_hh rows live in per-lane registers (outputs j and j+32);
// hidden state (20 floats) lives in LDS. Ends with z0 = hT @ W_rl^T + b_rl.
// Weight preloads use clamped indices + 0/1 masks (no exec-branch chains).
// ---------------------------------------------------------------------------
__global__ __launch_bounds__(256) void k_gru(const float* __restrict__ xW,
                                             const float* __restrict__ W_hh,
                                             const float* __restrict__ b_hh,
                                             const float* __restrict__ W_rl,
                                             const float* __restrict__ b_rl,
                                             float* __restrict__ z0out) {
    __shared__ float sh_h[8][RHID];
    __shared__ float sh_g[8][64];
    const int w = threadIdx.x >> 5;
    const int j = threadIdx.x & 31;
    const int b = blockIdx.x * 8 + w;

    const int   jj = (j < 28) ? (j + 32) : 59;     // clamped safe row
    const float m2 = (j < 28) ? 1.f : 0.f;
    float W1[RHID], W2[RHID];
    const float b1 = b_hh[j];
    const float b2 = b_hh[jj] * m2;
    #pragma unroll
    for (int i = 0; i < RHID; ++i) {
        W1[i] = W_hh[j  * RHID + i];
        W2[i] = W_hh[jj * RHID + i] * m2;
    }
    if (j < RHID) sh_h[w][j] = 0.f;
    __syncthreads();

    const float* xwb = xW + (size_t)b * T_IN * 64;
    for (int t = 0; t < T_IN; ++t) {
        float a1 = b1, a2 = b2;
        #pragma unroll
        for (int i = 0; i < RHID; ++i) {           // gh = h @ W_hh^T + b_hh
            const float hv = sh_h[w][i];
            a1 += W1[i] * hv;
            a2 += W2[i] * hv;
        }
        sh_g[w][j]      = a1;
        sh_g[w][j + 32] = a2;                      // slots 60..63 = dead pad
        __syncthreads();
        if (j < RHID) {
            const float* xwr = xwb + (size_t)t * 64;
            const float rg = 1.f / (1.f + __expf(-(xwr[j]          + sh_g[w][j])));
            const float zg = 1.f / (1.f + __expf(-(xwr[RHID + j]   + sh_g[w][RHID + j])));
            const float ng = tanhf(xwr[2 * RHID + j] + rg * sh_g[w][2 * RHID + j]);
            sh_h[w][j] = (1.f - zg) * ng + zg * sh_h[w][j];
        }
        __syncthreads();
    }
    if (j < LATENT) {
        float acc = b_rl[j];
        #pragma unroll
        for (int i = 0; i < RHID; ++i) acc += W_rl[j * RHID + i] * sh_h[w][i];
        z0out[b * LATENT + j] = acc;
    }
}

// ---------------------------------------------------------------------------
// Kernel 3: fixed-step RK4 latent ODE, 1023 steps. One wave per batch element
// (256 independent waves). MLP weights register-resident per lane (lane j
// owns hidden unit j); activations broadcast with __shfl. Writes pred_z.
// ---------------------------------------------------------------------------
__device__ __forceinline__ float elu1(float x) { return x > 0.f ? x : (__expf(x) - 1.f); }

__global__ __launch_bounds__(32) void k_rk4(const float* __restrict__ z0,
    const float* __restrict__ Wf1, const float* __restrict__ bf1,
    const float* __restrict__ Wf2, const float* __restrict__ bf2,
    const float* __restrict__ Wf3, const float* __restrict__ bf3,
    const float* __restrict__ Wf4, const float* __restrict__ bf4,
    const float* __restrict__ Wf5, const float* __restrict__ bf5,
    float* __restrict__ predz) {
    const int   b  = blockIdx.x;
    const int   j  = threadIdx.x;
    const int   jh = (j < NHID)   ? j : 0;         // clamped indices
    const int   jl = (j < LATENT) ? j : 0;
    const float mh = (j < NHID)   ? 1.f : 0.f;     // masks
    const float ml = (j < LATENT) ? 1.f : 0.f;

    float w1[LATENT], w2[NHID], w3[NHID], w4[NHID], w5[NHID];
    #pragma unroll
    for (int i = 0; i < LATENT; ++i) w1[i] = Wf1[jh * LATENT + i] * mh;
    #pragma unroll
    for (int i = 0; i < NHID; ++i) {
        w2[i] = Wf2[jh * NHID + i] * mh;
        w3[i] = Wf3[jh * NHID + i] * mh;
        w4[i] = Wf4[jh * NHID + i] * mh;
        w5[i] = Wf5[jl * NHID + i] * ml;
    }
    const float c1 = bf1[jh] * mh, c2 = bf2[jh] * mh;
    const float c3 = bf3[jh] * mh, c4 = bf4[jh] * mh;
    const float c5 = bf5[jl] * ml;

    auto fEval = [&](float zv) -> float {          // lanes 0..3 carry z/out
        const float zs0 = __shfl(zv, 0), zs1 = __shfl(zv, 1);
        const float zs2 = __shfl(zv, 2), zs3 = __shfl(zv, 3);
        float h = elu1(c1 + w1[0]*zs0 + w1[1]*zs1 + w1[2]*zs2 + w1[3]*zs3);
        float h2 = c2;
        #pragma unroll
        for (int i = 0; i < NHID; ++i) h2 += w2[i] * __shfl(h, i);
        h2 = elu1(h2);
        float h3 = c3;
        #pragma unroll
        for (int i = 0; i < NHID; ++i) h3 += w3[i] * __shfl(h2, i);
        h3 = elu1(h3);
        float h4 = c4;
        #pragma unroll
        for (int i = 0; i < NHID; ++i) h4 += w4[i] * __shfl(h3, i);
        h4 = elu1(h4);
        float o = c5;
        #pragma unroll
        for (int i = 0; i < NHID; ++i) o += w5[i] * __shfl(h4, i);
        return o;
    };

    float z = z0[b * LATENT + jl] * ml;
    if (j < LATENT) predz[(size_t)b * T_OUT * LATENT + j] = z;

    const float dt = 0.01f;
    for (int t = 1; t < T_OUT; ++t) {
        const float k1 = fEval(z);
        const float k2 = fEval(z + 0.5f * dt * k1);
        const float k3 = fEval(z + 0.5f * dt * k2);
        const float k4 = fEval(z + dt * k3);
        z = z + (dt / 6.f) * (k1 + 2.f * k2 + 2.f * k3 + k4);
        if (j < LATENT) predz[((size_t)b * T_OUT + t) * LATENT + j] = z;
    }
}

// ---------------------------------------------------------------------------
// Kernel 4: decoder. Stage 1: relu(pred_z @ W_d1^T + b_d1) via WMMA (K=4
// padded into one 16x16x32). Relayout 16x20 (stored 16x32, pad cols exactly
// zero) through per-wave LDS into the stage-2 A fragment. Stage 2: @ W_d2^T
// (K=20 padded to 32, N=192 -> 12 WMMA tiles), bias, scatter to [B,3,T,64].
// All LDS weight tiles padded -> every fragment load is contiguous+aligned.
// ---------------------------------------------------------------------------
__global__ __launch_bounds__(256) void k_decoder(const float* __restrict__ predz,
                                                 const float* __restrict__ Wd1,
                                                 const float* __restrict__ bd1,
                                                 const float* __restrict__ Wd2,
                                                 const float* __restrict__ bd2,
                                                 float* __restrict__ out) {
    __shared__ _Float16 wd2p[OBS * 32];            // [192][32], K pad 20->32
    __shared__ _Float16 wd1p[32 * 32];             // [32][32], N pad, K pad 4->32
    __shared__ float    bd1p[32];
    __shared__ float    bd2s[OBS];
    __shared__ _Float16 hbuf[8][16 * 32];          // per-wave 16x32 activations

    for (int i = threadIdx.x; i < OBS * 32; i += 256) {
        const int n = i >> 5, k = i & 31;
        wd2p[i] = (k < NHID) ? (_Float16)Wd2[n * NHID + k] : (_Float16)0.f;
    }
    for (int i = threadIdx.x; i < 32 * 32; i += 256) {
        const int n = i >> 5, k = i & 31;
        wd1p[i] = (n < NHID && k < LATENT) ? (_Float16)Wd1[n * LATENT + k]
                                           : (_Float16)0.f;
    }
    if (threadIdx.x < 32)
        bd1p[threadIdx.x] = (threadIdx.x < NHID) ? bd1[threadIdx.x] : 0.f;
    if (threadIdx.x < OBS) bd2s[threadIdx.x] = bd2[threadIdx.x];
    __syncthreads();

    const int w     = threadIdx.x >> 5;
    const int lane  = threadIdx.x & 31;
    const int rowl  = lane & 15;
    const int hi    = lane >> 4;
    const int mtile = blockIdx.x * 8 + w;
    const int r0    = mtile * 16 + rowl;

    // Stage 1 A fragment: K=4 live values (lane group 0 only), rest zero.
    v16h a1;
    #pragma unroll
    for (int u = 0; u < 16; ++u) a1[u] = (_Float16)0.f;
    {
        const float* zp = predz + (size_t)r0 * LATENT;
        if (hi == 0) {
            a1[0] = (_Float16)zp[0]; a1[1] = (_Float16)zp[1];
            a1[2] = (_Float16)zp[2]; a1[3] = (_Float16)zp[3];
        }
    }

    #pragma unroll
    for (int nt = 0; nt < 2; ++nt) {
        const int n = nt * 16 + rowl;              // 0..31, padded: valid
        const v16h bf = *(const v16h*)&wd1p[n * 32 + hi * 16];
        v8f zc = {};
        v8f hc = __builtin_amdgcn_wmma_f32_16x16x32_f16(
            false, a1, false, bf, (short)0, zc, false, false);
        const float bb = bd1p[n];                  // pad cols -> 0
        #pragma unroll
        for (int v = 0; v < 8; ++v) {
            // pad cols get relu(0+0)=0: store unconditionally
            hbuf[w][(v + hi * 8) * 32 + n] = (_Float16)fmaxf(hc[v] + bb, 0.f);
        }
    }
    __syncthreads();

    // Stage 2 A fragment: two contiguous 8-half groups from padded hbuf.
    v16h a2;
    {
        const _Float16* hrow = &hbuf[w][rowl * 32 + hi * 16];
        #pragma unroll
        for (int u = 0; u < 8; ++u) {
            a2[u]     = hrow[u];                   // K = hi*16 + u
            a2[8 + u] = hrow[8 + u];               // K = hi*16 + 8 + u
        }
    }

    #pragma unroll
    for (int nt = 0; nt < 12; ++nt) {
        const int  n  = nt * 16 + rowl;
        const v16h bf = *(const v16h*)&wd2p[n * 32 + hi * 16];
        v8f zc = {};
        v8f cacc = __builtin_amdgcn_wmma_f32_16x16x32_f16(
            false, a2, false, bf, (short)0, zc, false, false);
        const int   c    = n >> 6;
        const int   fcol = n & 63;
        const float bb   = bd2s[n];
        #pragma unroll
        for (int v = 0; v < 8; ++v) {
            const int rr = mtile * 16 + v + hi * 8;
            const int b  = rr >> 10;
            const int t  = rr & 1023;
            out[(((size_t)b * 3 + c) * T_OUT + t) * 64 + fcol] = cacc[v] + bb;
        }
    }
}

// ---------------------------------------------------------------------------
extern "C" void kernel_launch(void* const* d_in, const int* in_sizes, int n_in,
                              void* d_out, int out_size, void* d_ws, size_t ws_size,
                              hipStream_t stream) {
    const float* x    = (const float*)d_in[0];
    // d_in[1] = output_length (1024, fixed by harness)
    const float* W_ih = (const float*)d_in[2];
    const float* b_ih = (const float*)d_in[3];
    const float* W_hh = (const float*)d_in[4];
    const float* b_hh = (const float*)d_in[5];
    const float* W_rl = (const float*)d_in[6];
    const float* b_rl = (const float*)d_in[7];
    const float* W_f1 = (const float*)d_in[8];
    const float* b_f1 = (const float*)d_in[9];
    const float* W_f2 = (const float*)d_in[10];
    const float* b_f2 = (const float*)d_in[11];
    const float* W_f3 = (const float*)d_in[12];
    const float* b_f3 = (const float*)d_in[13];
    const float* W_f4 = (const float*)d_in[14];
    const float* b_f4 = (const float*)d_in[15];
    const float* W_f5 = (const float*)d_in[16];
    const float* b_f5 = (const float*)d_in[17];
    const float* W_d1 = (const float*)d_in[18];
    const float* b_d1 = (const float*)d_in[19];
    const float* W_d2 = (const float*)d_in[20];
    const float* b_d2 = (const float*)d_in[21];

    // workspace layout (f32): xW [131072*64] | z0 [1024] | pred_z [256*1024*4]
    float* xW    = (float*)d_ws;
    float* z0    = xW + (size_t)NB * T_IN * 64;
    float* predz = z0 + (size_t)NB * LATENT;

    // 1) input-gate GEMM: 8192 M-tiles, 8 waves/block
    k_xw_gemm<<<1024, 256, 0, stream>>>(x, W_ih, b_ih, xW);
    // 2) GRU scan: 8 batch elements per block
    k_gru<<<NB / 8, 256, 0, stream>>>(xW, W_hh, b_hh, W_rl, b_rl, z0);
    // 3) RK4 scan: one wave per batch element
    k_rk4<<<NB, 32, 0, stream>>>(z0, W_f1, b_f1, W_f2, b_f2, W_f3, b_f3,
                                 W_f4, b_f4, W_f5, b_f5, predz);
    // 4) decoder GEMM: 16384 M-tiles, 8 waves/block
    k_decoder<<<2048, 256, 0, stream>>>(predz, W_d1, b_d1, W_d2, b_d2,
                                        (float*)d_out);
}
